// GINLayer_50397146251358
// MI455X (gfx1250) — compile-verified, matrix operand order checked by reading, not run on hardware
//
#include <hip/hip_runtime.h>
#include <hip/hip_bf16.h>

typedef _Float16 v16h __attribute__((ext_vector_type(16)));
typedef _Float16 v8h  __attribute__((ext_vector_type(8)));
typedef _Float16 v4h  __attribute__((ext_vector_type(4)));
typedef float    v8f  __attribute__((ext_vector_type(8)));

#define N_NODES 100000
#define N_EDGES 320000
#define D_IN    256
#define D_HID   1024
#define D_OUT   256
#define BN_EPS  1e-5f

// LDS tile: 128 rows x 32 halves, padded to stride 40 halves (80B = 20 banks)
// so the 16-lane x 16B fragment reads are bank-conflict free.
#define LDT 40

// async global -> LDS, 16 bytes, GV mode (64-bit vaddr, no saddr)
#define ASYNC_LD_B128(ldsOff, gptr)                                             \
    asm volatile("global_load_async_to_lds_b128 %0, %1, off"                    \
                 :: "v"(ldsOff), "v"(gptr) : "memory")
#define WAIT_ASYNC0() asm volatile("s_wait_asynccnt 0x0" ::: "memory")

// ---------------------------------------------------------------------------
// Elementwise / prep kernels
// ---------------------------------------------------------------------------

__global__ void k_copy_f32x4(const float* __restrict__ in, float* __restrict__ out, int n4) {
    int idx = blockIdx.x * blockDim.x + threadIdx.x;
    if (idx < n4) ((float4*)out)[idx] = ((const float4*)in)[idx];
}

// edge scatter: h[dst] += x[src], 64 threads per edge, 4 floats per thread
__global__ void k_scatter(const float* __restrict__ x, const int* __restrict__ src,
                          const int* __restrict__ dst, float* __restrict__ h, int nEdges) {
    int idx = blockIdx.x * blockDim.x + threadIdx.x;
    int e = idx >> 6;
    if (e >= nEdges) return;
    int j = (idx & 63) * 4;
    int s = src[e], d = dst[e];
    float4 v = *(const float4*)(x + (size_t)s * D_IN + j);
    float* p = h + (size_t)d * D_IN + j;
    atomicAdd(p + 0, v.x);
    atomicAdd(p + 1, v.y);
    atomicAdd(p + 2, v.z);
    atomicAdd(p + 3, v.w);
}

__global__ void k_f32_to_f16x4(const float* __restrict__ in, _Float16* __restrict__ out, int n4) {
    int idx = blockIdx.x * blockDim.x + threadIdx.x;
    if (idx >= n4) return;
    float4 v = ((const float4*)in)[idx];
    v4h o;
    o[0] = (_Float16)v.x; o[1] = (_Float16)v.y; o[2] = (_Float16)v.z; o[3] = (_Float16)v.w;
    ((v4h*)out)[idx] = o;
}

// dst[c*R + r] = (f16) src[r*C + c]   (produce N-major, K-contiguous weights)
__global__ void k_transpose_f16(const float* __restrict__ in, _Float16* __restrict__ out,
                                int R, int C) {
    int idx = blockIdx.x * blockDim.x + threadIdx.x;
    if (idx >= R * C) return;
    int r = idx / C, c = idx % C;
    out[(size_t)c * R + r] = (_Float16)in[idx];
}

__global__ void k_zero_f32(float* __restrict__ p, int n) {
    int idx = blockIdx.x * blockDim.x + threadIdx.x;
    if (idx < n) p[idx] = 0.0f;
}

// column sums over h1[M][1024] (f16): blockDim = 256, each thread owns 4 channels
__global__ void k_bnstats(const _Float16* __restrict__ h1, float* __restrict__ sum,
                          float* __restrict__ sumsq, int M) {
    int c0 = threadIdx.x * 4;
    float s0 = 0.f, s1 = 0.f, s2 = 0.f, s3 = 0.f;
    float q0 = 0.f, q1 = 0.f, q2 = 0.f, q3 = 0.f;
    for (int r = blockIdx.x; r < M; r += gridDim.x) {
        v4h v = *(const v4h*)(h1 + (size_t)r * D_HID + c0);
        float f0 = (float)v[0], f1 = (float)v[1], f2 = (float)v[2], f3 = (float)v[3];
        s0 += f0; s1 += f1; s2 += f2; s3 += f3;
        q0 += f0 * f0; q1 += f1 * f1; q2 += f2 * f2; q3 += f3 * f3;
    }
    atomicAdd(&sum[c0 + 0], s0); atomicAdd(&sum[c0 + 1], s1);
    atomicAdd(&sum[c0 + 2], s2); atomicAdd(&sum[c0 + 3], s3);
    atomicAdd(&sumsq[c0 + 0], q0); atomicAdd(&sumsq[c0 + 1], q1);
    atomicAdd(&sumsq[c0 + 2], q2); atomicAdd(&sumsq[c0 + 3], q3);
}

__global__ void k_bnfinal(const float* __restrict__ sum, const float* __restrict__ sumsq,
                          const float* __restrict__ gamma, const float* __restrict__ beta,
                          float* __restrict__ scale, float* __restrict__ shift, int M) {
    int c = blockIdx.x * blockDim.x + threadIdx.x;
    if (c >= D_HID) return;
    float invM = 1.0f / (float)M;
    float mean = sum[c] * invM;
    float var  = sumsq[c] * invM - mean * mean;
    float inv  = rsqrtf(var + BN_EPS);
    float sc   = gamma[c] * inv;
    scale[c] = sc;
    shift[c] = beta[c] - mean * sc;
}

// in-place BN + ReLU on h1 (f16), 8 halves per thread
__global__ void k_bnapply(_Float16* __restrict__ h1, const float* __restrict__ scale,
                          const float* __restrict__ shift, int n8) {
    int idx = blockIdx.x * blockDim.x + threadIdx.x;
    if (idx >= n8) return;
    int c0 = (idx * 8) & (D_HID - 1);
    v8h v = ((v8h*)h1)[idx];
#pragma unroll
    for (int i = 0; i < 8; ++i) {
        float f = (float)v[i] * scale[c0 + i] + shift[c0 + i];
        v[i] = (_Float16)fmaxf(f, 0.0f);
    }
    ((v8h*)h1)[idx] = v;
}

// ---------------------------------------------------------------------------
// WMMA GEMM: C[M][N] = A[M][K] * B^T (Bt is [N][K], K-contiguous) + bias
// Block: 128x128 tile, 256 threads = 8 waves, wave = 32x64 (2x4 WMMA tiles).
// Double-buffered LDS, tiles filled by GLOBAL_LOAD_ASYNC_TO_LDS_B128.
// ---------------------------------------------------------------------------

// Fragment per the CDNA5 16-bit A-matrix layout: lane -> row(0..15),
// lane half selects K sub-run: VGPR0-3 hold K = 8*(lane>>4)+0..7,
// VGPR4-7 hold K = 16 + 8*(lane>>4)+0..7. (B uses the same layout on Bt.)
__device__ __forceinline__ v16h load_frag(const _Float16* __restrict__ tile, int rowBase, int lane) {
    const int r  = rowBase + (lane & 15);
    const int kh = (lane >> 4) * 8;
    v8h lo = *(const v8h*)(tile + r * LDT + kh);
    v8h hi = *(const v8h*)(tile + r * LDT + kh + 16);
    v16h out;
#pragma unroll
    for (int i = 0; i < 8; ++i) { out[i] = lo[i]; out[i + 8] = hi[i]; }
    return out;
}

template <int EPI>  // 0: +bias, store f16 ; 1: +bias, relu, store f32
__global__ __launch_bounds__(256) void gemm_wmma(
    const _Float16* __restrict__ A, const _Float16* __restrict__ Bt,
    const float* __restrict__ bias, void* __restrict__ Cout,
    int M, int N, int K)
{
    __shared__ _Float16 sA[2][128 * LDT];
    __shared__ _Float16 sB[2][128 * LDT];

    const int tid  = threadIdx.x;
    const int lane = tid & 31;
    const int wave = tid >> 5;
    const int wm   = (wave & 3) * 32;   // wave row offset in block
    const int wn   = (wave >> 2) * 64;  // wave col offset in block
    const int blockM = blockIdx.x * 128;
    const int blockN = blockIdx.y * 128;

    // ---- staging geometry: 512 x 16B chunks, 2 per thread per tile ----
    const int r0 = tid >> 2;                 // rows 0..63
    const int r1 = r0 + 64;                  // rows 64..127
    const int c  = (tid & 3) * 8;            // K offset in halves (0,8,16,24)
    int gaR0 = blockM + r0; if (gaR0 >= M) gaR0 = M - 1;
    int gaR1 = blockM + r1; if (gaR1 >= M) gaR1 = M - 1;
    const _Float16* pa0 = A  + (size_t)gaR0 * K + c;
    const _Float16* pa1 = A  + (size_t)gaR1 * K + c;
    const _Float16* pb0 = Bt + (size_t)(blockN + r0) * K + c;   // N % 128 == 0
    const _Float16* pb1 = Bt + (size_t)(blockN + r1) * K + c;

    const unsigned offL0 = (unsigned)(r0 * LDT + c) * 2u;       // LDS byte offsets
    const unsigned offL1 = (unsigned)(r1 * LDT + c) * 2u;
    unsigned baseA[2], baseB[2];
    baseA[0] = (unsigned)(size_t)&sA[0][0];
    baseA[1] = (unsigned)(size_t)&sA[1][0];
    baseB[0] = (unsigned)(size_t)&sB[0][0];
    baseB[1] = (unsigned)(size_t)&sB[1][0];

    const int nk = K >> 5;

    // prologue: fill buffer 0
    ASYNC_LD_B128(baseA[0] + offL0, pa0);
    ASYNC_LD_B128(baseA[0] + offL1, pa1);
    ASYNC_LD_B128(baseB[0] + offL0, pb0);
    ASYNC_LD_B128(baseB[0] + offL1, pb1);

    v8f acc[2][4];
#pragma unroll
    for (int mt = 0; mt < 2; ++mt)
#pragma unroll
        for (int nt = 0; nt < 4; ++nt)
#pragma unroll
            for (int e = 0; e < 8; ++e) acc[mt][nt][e] = 0.0f;

    for (int ki = 0; ki < nk; ++ki) {
        WAIT_ASYNC0();        // this thread's fills of buffer (ki&1) have landed
        __syncthreads();      // everyone's fills landed; previous reads done

        if (ki + 1 < nk) {    // prefetch next K-slab into the other buffer
            const int b  = (ki + 1) & 1;
            const int kO = (ki + 1) * 32;
            ASYNC_LD_B128(baseA[b] + offL0, pa0 + kO);
            ASYNC_LD_B128(baseA[b] + offL1, pa1 + kO);
            ASYNC_LD_B128(baseB[b] + offL0, pb0 + kO);
            ASYNC_LD_B128(baseB[b] + offL1, pb1 + kO);
        }

        const _Float16* tA = sA[ki & 1];
        const _Float16* tB = sB[ki & 1];
        v16h af[2], bf[4];
#pragma unroll
        for (int mt = 0; mt < 2; ++mt) af[mt] = load_frag(tA, wm + mt * 16, lane);
#pragma unroll
        for (int nt = 0; nt < 4; ++nt) bf[nt] = load_frag(tB, wn + nt * 16, lane);

#pragma unroll
        for (int mt = 0; mt < 2; ++mt)
#pragma unroll
            for (int nt = 0; nt < 4; ++nt)
                acc[mt][nt] = __builtin_amdgcn_wmma_f32_16x16x32_f16(
                    false, af[mt], false, bf[nt], (short)0, acc[mt][nt], false, false);
    }

    // ---- epilogue -----------------------------------------------------------
    // C/D layout: lane -> n (0..15), VGPR v -> m = v + 8*(lane>>4)
    const int nLane = lane & 15;
    const int mOff  = (lane >> 4) * 8;
    const int gmBase = blockM + wm + mOff;

    if (blockM + 128 <= M) {            // full tile: no predicates
#pragma unroll
        for (int nt = 0; nt < 4; ++nt) {
            const int gn = blockN + wn + nt * 16 + nLane;
            const float bv = bias[gn];
#pragma unroll
            for (int mt = 0; mt < 2; ++mt) {
                if (EPI == 1) {
                    float* p = (float*)Cout + (size_t)(gmBase + mt * 16) * N + gn;
#pragma unroll
                    for (int v = 0; v < 8; ++v)
                        p[(size_t)v * N] = fmaxf(acc[mt][nt][v] + bv, 0.0f);
                } else {
                    _Float16* p = (_Float16*)Cout + (size_t)(gmBase + mt * 16) * N + gn;
#pragma unroll
                    for (int v = 0; v < 8; ++v)
                        p[(size_t)v * N] = (_Float16)(acc[mt][nt][v] + bv);
                }
            }
        }
    } else {                            // ragged last M-block
#pragma unroll
        for (int nt = 0; nt < 4; ++nt) {
            const int gn = blockN + wn + nt * 16 + nLane;
            const float bv = bias[gn];
#pragma unroll
            for (int mt = 0; mt < 2; ++mt) {
#pragma unroll
                for (int v = 0; v < 8; ++v) {
                    const int gm = gmBase + mt * 16 + v;
                    if (gm < M) {
                        float val = acc[mt][nt][v] + bv;
                        if (EPI == 1)
                            ((float*)Cout)[(size_t)gm * N + gn] = fmaxf(val, 0.0f);
                        else
                            ((_Float16*)Cout)[(size_t)gm * N + gn] = (_Float16)val;
                    }
                }
            }
        }
    }
}

// ---------------------------------------------------------------------------
// Launch
// ---------------------------------------------------------------------------

extern "C" void kernel_launch(void* const* d_in, const int* in_sizes, int n_in,
                              void* d_out, int out_size, void* d_ws, size_t ws_size,
                              hipStream_t stream) {
    (void)in_sizes; (void)n_in; (void)out_size; (void)ws_size;

    const float* x     = (const float*)d_in[0];
    const int*   eidx  = (const int*)d_in[1];   // [2][N_EDGES]
    const float* W1    = (const float*)d_in[2];
    const float* b1    = (const float*)d_in[3];
    const float* gamma = (const float*)d_in[4];
    const float* beta  = (const float*)d_in[5];
    const float* W2    = (const float*)d_in[6];
    const float* b2    = (const float*)d_in[7];
    float* out = (float*)d_out;

    char* ws = (char*)d_ws;
    size_t off = 0;
    auto alloc = [&](size_t bytes) -> char* {
        char* p = ws + off;
        off += (bytes + 255) & ~(size_t)255;
        return p;
    };

    // region0 is used first as h (f32, 102.4MB), then reused as h1 (f16, 204.8MB)
    char*     region0 = alloc((size_t)N_NODES * D_HID * sizeof(_Float16));
    float*    hF32 = (float*)region0;
    _Float16* h1   = (_Float16*)region0;
    _Float16* hA   = (_Float16*)alloc((size_t)N_NODES * D_IN * sizeof(_Float16));
    _Float16* W1t  = (_Float16*)alloc((size_t)D_IN * D_HID * sizeof(_Float16));
    _Float16* W2t  = (_Float16*)alloc((size_t)D_HID * D_OUT * sizeof(_Float16));
    float*    sums  = (float*)alloc(D_HID * sizeof(float));
    float*    sumsq = (float*)alloc(D_HID * sizeof(float));
    float*    scale = (float*)alloc(D_HID * sizeof(float));
    float*    shift = (float*)alloc(D_HID * sizeof(float));

    // 1. weight transpose + f16 convert (Bt layout: [N][K], K contiguous)
    k_transpose_f16<<<(D_IN * D_HID + 255) / 256, 256, 0, stream>>>(W1, W1t, D_IN, D_HID);
    k_transpose_f16<<<(D_HID * D_OUT + 255) / 256, 256, 0, stream>>>(W2, W2t, D_HID, D_OUT);

    // 2. h = x ; h[dst] += x[src] ; hA = f16(h)
    const int nH4 = N_NODES * D_IN / 4;
    k_copy_f32x4<<<(nH4 + 255) / 256, 256, 0, stream>>>(x, hF32, nH4);
    const int scatterThreads = N_EDGES * 64;
    k_scatter<<<(scatterThreads + 255) / 256, 256, 0, stream>>>(
        x, eidx, eidx + N_EDGES, hF32, N_EDGES);
    k_f32_to_f16x4<<<(nH4 + 255) / 256, 256, 0, stream>>>(hF32, hA, nH4);

    // 3. h1 = hA @ W1 + b1  (f16 out)   -- overwrites region0, hF32 now dead
    {
        dim3 grid((N_NODES + 127) / 128, D_HID / 128);
        gemm_wmma<0><<<grid, 256, 0, stream>>>(hA, W1t, b1, (void*)h1,
                                               N_NODES, D_HID, D_IN);
    }

    // 4. batch-norm statistics + finalize + apply (with ReLU) in place
    k_zero_f32<<<(2 * D_HID + 255) / 256, 256, 0, stream>>>(sums, 2 * D_HID);
    k_bnstats<<<480, 256, 0, stream>>>(h1, sums, sumsq, N_NODES);
    k_bnfinal<<<(D_HID + 255) / 256, 256, 0, stream>>>(sums, sumsq, gamma, beta,
                                                       scale, shift, N_NODES);
    const int n8 = N_NODES * D_HID / 8;
    k_bnapply<<<(n8 + 255) / 256, 256, 0, stream>>>(h1, scale, shift, n8);

    // 5. out = relu(h1 @ W2 + b2)  (f32 out)
    {
        dim3 grid((N_NODES + 127) / 128, D_OUT / 128);
        gemm_wmma<1><<<grid, 256, 0, stream>>>(h1, W2t, b2, (void*)out,
                                               N_NODES, D_OUT, D_HID);
    }
}
